// CrossAttLayer_34926674051617
// MI455X (gfx1250) — matrix operand order repaired and checked by reading, not compiled, not run on hardware
//
#include <hip/hip_runtime.h>

// ---------------------------------------------------------------------------
// CrossAttLayer for MI455X (gfx1250, wave32, WMMA)
//   Phase 1: node-level MLPs (factored out of per-edge loop: 32x fewer FLOPs)
//            via v_wmma_f32_16x16x32_bf16, 16x64 strip per wave.
//   Phase 2: CSR build (int atomics only) + fused per-node ONLINE-softmax
//            aggregate wave kernel — single pass over edges, zero float
//            atomics, L2-resident gathers + global_prefetch_b8.
// ---------------------------------------------------------------------------

typedef __attribute__((ext_vector_type(16))) __bf16 v16bf;
typedef __attribute__((ext_vector_type(8)))  __bf16 v8bf;
typedef __attribute__((ext_vector_type(8)))  float  v8f;

#define OUT_DIM   128
#define HID_DIM   256
#define N_HEADS   8
#define HEAD_DIM  16

// ---- helpers ---------------------------------------------------------------

__device__ __forceinline__ unsigned short f2bf(float f) {
  // round-to-nearest-even f32 -> bf16
  unsigned int u = __float_as_uint(f);
  unsigned int r = u + 0x7FFFu + ((u >> 16) & 1u);
  return (unsigned short)(r >> 16);
}

// ---- tiny utility kernels --------------------------------------------------

__global__ __launch_bounds__(256) void zero_u32_kernel(unsigned int* p, int n) {
  int i = blockIdx.x * blockDim.x + threadIdx.x;
  if (i < n) p[i] = 0u;
}

__global__ __launch_bounds__(256) void f32_to_bf16_kernel(const float* __restrict__ x,
                                                          unsigned short* __restrict__ y,
                                                          int n) {
  int i = blockIdx.x * blockDim.x + threadIdx.x;
  if (i < n) y[i] = f2bf(x[i]);
}

__global__ __launch_bounds__(256) void build_inv_kernel(const int* __restrict__ center,
                                                        int* __restrict__ inv, int n) {
  int i = blockIdx.x * blockDim.x + threadIdx.x;
  if (i < n) inv[center[i]] = i;
}

// Pack W[K,N] (row-major f32) into per-wave WMMA B fragments (bf16):
//   frag index t = kt*(N/16)+nt ; lane l<16 holds col nt*16+l, K=kt*32+0..15,
//   lane l>=16 holds col nt*16+(l-16), K=kt*32+16..31; 16 contiguous halfs/lane.
__global__ __launch_bounds__(256) void pack_b_kernel(const float* __restrict__ W,
                                                     unsigned short* __restrict__ Bp,
                                                     int K, int N) {
  int idx = blockIdx.x * blockDim.x + threadIdx.x;      // one per element
  if (idx >= K * N) return;
  int j = idx & 15;
  int l = (idx >> 4) & 31;
  int t = idx >> 9;
  int tiles_n = N >> 4;
  int kt = t / tiles_n;
  int nt = t - kt * tiles_n;
  int n = nt * 16 + (l & 15);
  int k = kt * 32 + (l >> 4) * 16 + j;
  Bp[idx] = f2bf(W[(size_t)k * N + n]);
}

// ---- CSR build over dst_new -----------------------------------------------

__global__ __launch_bounds__(256) void count_kernel(const int* __restrict__ dst,
                                                    const int* __restrict__ inv,
                                                    int* __restrict__ counts, int E) {
  int e = blockIdx.x * blockDim.x + threadIdx.x;
  if (e < E) atomicAdd(&counts[inv[dst[e]]], 1);
}

// single-block exclusive scan: row_start[0..N], row_start[N] = E
__global__ __launch_bounds__(256) void scan_kernel(const int* __restrict__ counts,
                                                   int* __restrict__ row_start, int N) {
  __shared__ int buf[256];
  __shared__ int carry_s;
  if (threadIdx.x == 0) carry_s = 0;
  __syncthreads();
  for (int base = 0; base < N; base += 256) {
    int i = base + threadIdx.x;
    int v = (i < N) ? counts[i] : 0;
    buf[threadIdx.x] = v;
    __syncthreads();
    // Hillis-Steele inclusive scan
    for (int off = 1; off < 256; off <<= 1) {
      int t = (threadIdx.x >= off) ? buf[threadIdx.x - off] : 0;
      __syncthreads();
      buf[threadIdx.x] += t;
      __syncthreads();
    }
    int carry = carry_s;
    if (i < N) row_start[i] = carry + buf[threadIdx.x] - v;  // exclusive
    int total = buf[255];
    __syncthreads();
    if (threadIdx.x == 0) carry_s = carry + total;
    __syncthreads();
  }
  if (threadIdx.x == 0) row_start[N] = carry_s;
}

__global__ __launch_bounds__(256) void scatter_kernel(
    const int* __restrict__ src, const int* __restrict__ dst,
    const int* __restrict__ inv, const int* __restrict__ row_start,
    int* __restrict__ cursor, int* __restrict__ esrc, int E) {
  int e = blockIdx.x * blockDim.x + threadIdx.x;
  if (e >= E) return;
  int dn = inv[dst[e]];
  int p = atomicAdd(&cursor[dn], 1);
  esrc[row_start[dn] + p] = src[e];
}

// ---- WMMA GEMM: C = act(A_bf16[M,K] @ W + bias) ----------------------------
// One wave computes a 16x64 strip (4 accumulators: hides the 5-slot
// WMMA->WMMA RAW hazard and reuses the A fragment 4x). K consumed 32/step.

template <bool RELU_BF16_OUT>
__global__ __launch_bounds__(256) void gemm_wmma_kernel(
    const unsigned short* __restrict__ A,   // bf16 [M,K] row-major
    const unsigned short* __restrict__ Bp,  // packed bf16 fragments
    const float* __restrict__ bias,         // [N]
    void* __restrict__ C, int M, int N, int K) {
  const int wave = threadIdx.x >> 5;            // wave32
  const int lane = threadIdx.x & 31;
  const int tiles_n = N >> 4;
  const int groups_n = N >> 6;                  // 4 tiles per group
  const int group = blockIdx.x * (blockDim.x >> 5) + wave;
  if (group >= (M >> 4) * groups_n) return;     // wave-uniform guard
  const int tm = group / groups_n;
  const int tn0 = (group - tm * groups_n) * 4;

  const int lo = lane & 15;
  const int hi = lane >> 4;                     // 0 or 1
  // A 16x32 fragment, contiguous k (as packed B): lanes 0-15 -> row M=lo, K = 0..15;
  //                        lanes 16-31 -> row M=lo, K = 16..31
  const unsigned short* arow = A + (size_t)(tm * 16 + lo) * K;

  v8f acc0 = {}, acc1 = {}, acc2 = {}, acc3 = {};
  for (int kk = 0; kk < K; kk += 32) {
    const int ka = kk + hi * 16;
    v8bf alo = *(const v8bf*)(arow + ka);         // 16B contiguous
    v8bf ahi = *(const v8bf*)(arow + ka + 8);     // 16B contiguous
    v16bf af = __builtin_shufflevector(alo, ahi, 0, 1, 2, 3, 4, 5, 6, 7,
                                       8, 9, 10, 11, 12, 13, 14, 15);
    const unsigned short* bbase =
        Bp + ((size_t)((kk >> 5) * tiles_n + tn0) * 32 + lane) * 16;
    v16bf b0 = *(const v16bf*)(bbase);
    v16bf b1 = *(const v16bf*)(bbase + 512);      // next 16-col tile (32*16)
    v16bf b2 = *(const v16bf*)(bbase + 1024);
    v16bf b3 = *(const v16bf*)(bbase + 1536);
    acc0 = __builtin_amdgcn_wmma_f32_16x16x32_bf16(false, af, false, b0, (short)0, acc0, false, false);
    acc1 = __builtin_amdgcn_wmma_f32_16x16x32_bf16(false, af, false, b1, (short)0, acc1, false, false);
    acc2 = __builtin_amdgcn_wmma_f32_16x16x32_bf16(false, af, false, b2, (short)0, acc2, false, false);
    acc3 = __builtin_amdgcn_wmma_f32_16x16x32_bf16(false, af, false, b3, (short)0, acc3, false, false);
  }

  // 32-bit C/D layout: lane<16 -> col=lane, rows r; lane>=16 -> col=lane-16, rows 8+r
  const int r0 = tm * 16 + hi * 8;
  v8f* accs[4] = { &acc0, &acc1, &acc2, &acc3 };
#pragma unroll
  for (int j = 0; j < 4; ++j) {
    const int col = (tn0 + j) * 16 + lo;
    const float bv = bias[col];
    if (RELU_BF16_OUT) {
      unsigned short* cp = (unsigned short*)C;
#pragma unroll
      for (int r = 0; r < 8; ++r) {
        float v = (*accs[j])[r] + bv;
        v = v > 0.f ? v : 0.f;
        cp[(size_t)(r0 + r) * N + col] = f2bf(v);
      }
    } else {
      float* cp = (float*)C;
#pragma unroll
      for (int r = 0; r < 8; ++r)
        cp[(size_t)(r0 + r) * N + col] = (*accs[j])[r] + bv;
    }
  }
}

// ---- fused per-node ONLINE softmax + aggregation ---------------------------
// One wave per destination node. Lanes = 8 head-groups x 4 lanes; each lane
// owns 4 of the 16 head dims. Single pass with running (max, sum, acc) and
// rescale-on-new-max (flash style): denominator applied once at the end.
// K+V read exactly once per edge; next edge's rows prefetched.

__global__ __launch_bounds__(256) void node_attn_kernel(
    const float* __restrict__ Q, const float* __restrict__ Kf,
    const float* __restrict__ V, const int* __restrict__ center,
    const int* __restrict__ row_start, const int* __restrict__ esrc,
    float* __restrict__ out, int N) {
  const int wave = threadIdx.x >> 5;
  const int lane = threadIdx.x & 31;
  const int n = blockIdx.x * (blockDim.x >> 5) + wave;
  if (n >= N) return;
  const int head = lane >> 2;
  const int sub = lane & 3;
  const int off = head * HEAD_DIM + sub * 4;

  const int qn = center[n];                       // out row n <=> dst node center[n]
  const float4 q4 = *(const float4*)(Q + (size_t)qn * OUT_DIM + off);
  const int beg = row_start[n], end = row_start[n + 1];

  float m = -3.402823466e38f;                     // running max
  float sum = 0.f;                                // running exp-sum
  float ax = 0.f, ay = 0.f, az = 0.f, aw = 0.f;   // running weighted-V

  for (int i = beg; i < end; ++i) {
    const int s = esrc[i];
    if (i + 1 < end) {                            // hide L2 gather latency
      const int sn = esrc[i + 1];
      __builtin_prefetch(Kf + (size_t)sn * OUT_DIM + off, 0, 0);  // global_prefetch_b8
      __builtin_prefetch(V + (size_t)sn * OUT_DIM + off, 0, 0);
    }
    const float4 k4 = *(const float4*)(Kf + (size_t)s * OUT_DIM + off);
    const float4 v4 = *(const float4*)(V + (size_t)s * OUT_DIM + off);
    float d = q4.x * k4.x + q4.y * k4.y + q4.z * k4.z + q4.w * k4.w;
    d += __shfl_xor(d, 1, 32);                    // reduce across the 4-lane
    d += __shfl_xor(d, 2, 32);                    //   head group (wave32)
    d *= 0.25f;                                   // 1/sqrt(HEAD_DIM)

    const float mn = fmaxf(m, d);
    const float scale = __expf(m - mn);           // 1 if max unchanged; 0 on first edge
    const float w = __expf(d - mn);
    sum = sum * scale + w;
    ax = ax * scale + w * v4.x;
    ay = ay * scale + w * v4.y;
    az = az * scale + w * v4.z;
    aw = aw * scale + w * v4.w;
    m = mn;
  }

  const float r = (sum > 0.f) ? (1.f / sum) : 0.f;  // empty segment -> 0
  float4 o;
  o.x = ax * r; o.y = ay * r; o.z = az * r; o.w = aw * r;
  *(float4*)(out + (size_t)n * OUT_DIM + off) = o;
}

// ---------------------------------------------------------------------------

extern "C" void kernel_launch(void* const* d_in, const int* in_sizes, int n_in,
                              void* d_out, int out_size, void* d_ws, size_t ws_size,
                              hipStream_t stream) {
  const float* h      = (const float*)d_in[0];
  const float* a_mol  = (const float*)d_in[1];
  const int*   src    = (const int*)d_in[2];
  const int*   dst    = (const int*)d_in[3];
  const int*   center = (const int*)d_in[4];
  const float* Wk1 = (const float*)d_in[5];  const float* bk1 = (const float*)d_in[6];
  const float* Wk2 = (const float*)d_in[7];  const float* bk2 = (const float*)d_in[8];
  const float* Wv1 = (const float*)d_in[9];  const float* bv1 = (const float*)d_in[10];
  const float* Wv2 = (const float*)d_in[11]; const float* bv2 = (const float*)d_in[12];
  const float* Wq1 = (const float*)d_in[13]; const float* bq1 = (const float*)d_in[14];
  const float* Wq2 = (const float*)d_in[15]; const float* bq2 = (const float*)d_in[16];
  (void)n_in; (void)ws_size; (void)out_size;

  const int N = in_sizes[4];           // nodes (20000)
  const int E = in_sizes[2];           // edges (640000)
  const int IN_DIM = in_sizes[0] / N;  // 128
  float* out = (float*)d_out;

  // workspace carve-out (~55 MB)
  char* wp = (char*)d_ws;
  auto alloc = [&](size_t bytes) -> void* {
    void* p = (void*)wp;
    wp += (bytes + 255) & ~(size_t)255;
    return p;
  };
  unsigned short* h_bf = (unsigned short*)alloc((size_t)N * IN_DIM * 2);
  unsigned short* a_bf = (unsigned short*)alloc((size_t)N * IN_DIM * 2);
  unsigned short* act  = (unsigned short*)alloc((size_t)N * HID_DIM * 2);
  float* Qb = (float*)alloc((size_t)N * OUT_DIM * 4);
  float* Kb = (float*)alloc((size_t)N * OUT_DIM * 4);
  float* Vb = (float*)alloc((size_t)N * OUT_DIM * 4);
  int* inv       = (int*)alloc((size_t)N * 4);
  int* counts    = (int*)alloc((size_t)N * 4);
  int* cursor    = (int*)alloc((size_t)N * 4);
  int* row_start = (int*)alloc((size_t)(N + 1) * 4);
  int* esrc      = (int*)alloc((size_t)E * 4);
  unsigned short* Bk1 = (unsigned short*)alloc((size_t)IN_DIM * HID_DIM * 2);
  unsigned short* Bk2 = (unsigned short*)alloc((size_t)HID_DIM * OUT_DIM * 2);
  unsigned short* Bv1 = (unsigned short*)alloc((size_t)IN_DIM * HID_DIM * 2);
  unsigned short* Bv2 = (unsigned short*)alloc((size_t)HID_DIM * OUT_DIM * 2);
  unsigned short* Bq1 = (unsigned short*)alloc((size_t)IN_DIM * HID_DIM * 2);
  unsigned short* Bq2 = (unsigned short*)alloc((size_t)HID_DIM * OUT_DIM * 2);

  const int B = 256;
  auto gridOf = [](long n, int b) { return (int)((n + b - 1) / b); };

  // ---- CSR build (deterministic integer atomics) ----
  zero_u32_kernel<<<gridOf(N, B), B, 0, stream>>>((unsigned int*)counts, N);
  zero_u32_kernel<<<gridOf(N, B), B, 0, stream>>>((unsigned int*)cursor, N);
  build_inv_kernel<<<gridOf(N, B), B, 0, stream>>>(center, inv, N);
  count_kernel<<<gridOf(E, B), B, 0, stream>>>(dst, inv, counts, E);
  scan_kernel<<<1, B, 0, stream>>>(counts, row_start, N);
  scatter_kernel<<<gridOf(E, B), B, 0, stream>>>(src, dst, inv, row_start, cursor, esrc, E);

  // ---- precision conversion + weight packing ----
  f32_to_bf16_kernel<<<gridOf((long)N * IN_DIM, B), B, 0, stream>>>(h, h_bf, N * IN_DIM);
  f32_to_bf16_kernel<<<gridOf((long)N * IN_DIM, B), B, 0, stream>>>(a_mol, a_bf, N * IN_DIM);
  pack_b_kernel<<<gridOf((long)IN_DIM * HID_DIM, B), B, 0, stream>>>(Wk1, Bk1, IN_DIM, HID_DIM);
  pack_b_kernel<<<gridOf((long)HID_DIM * OUT_DIM, B), B, 0, stream>>>(Wk2, Bk2, HID_DIM, OUT_DIM);
  pack_b_kernel<<<gridOf((long)IN_DIM * HID_DIM, B), B, 0, stream>>>(Wv1, Bv1, IN_DIM, HID_DIM);
  pack_b_kernel<<<gridOf((long)HID_DIM * OUT_DIM, B), B, 0, stream>>>(Wv2, Bv2, HID_DIM, OUT_DIM);
  pack_b_kernel<<<gridOf((long)IN_DIM * HID_DIM, B), B, 0, stream>>>(Wq1, Bq1, IN_DIM, HID_DIM);
  pack_b_kernel<<<gridOf((long)HID_DIM * OUT_DIM, B), B, 0, stream>>>(Wq2, Bq2, HID_DIM, OUT_DIM);

  // ---- node-level MLPs via WMMA ----
  const int g1 = gridOf((long)(N / 16) * (HID_DIM / 64), 8);  // 16x64 strips
  const int g2 = gridOf((long)(N / 16) * (OUT_DIM / 64), 8);
  gemm_wmma_kernel<true ><<<g1, B, 0, stream>>>(a_bf, Bk1, bk1, act, N, HID_DIM, IN_DIM);
  gemm_wmma_kernel<false><<<g2, B, 0, stream>>>(act, Bk2, bk2, Kb, N, OUT_DIM, HID_DIM);
  gemm_wmma_kernel<true ><<<g1, B, 0, stream>>>(a_bf, Bv1, bv1, act, N, HID_DIM, IN_DIM);
  gemm_wmma_kernel<false><<<g2, B, 0, stream>>>(act, Bv2, bv2, Vb, N, OUT_DIM, HID_DIM);
  gemm_wmma_kernel<true ><<<g1, B, 0, stream>>>(h_bf, Bq1, bq1, act, N, HID_DIM, IN_DIM);
  gemm_wmma_kernel<false><<<g2, B, 0, stream>>>(act, Bq2, bq2, Qb, N, OUT_DIM, HID_DIM);

  // ---- fused per-node online-softmax attention (single pass, no atomics) ----
  node_attn_kernel<<<gridOf(N, 8), B, 0, stream>>>(Qb, Kb, Vb, center, row_start, esrc, out, N);
}
